// EISANImodel_44332652429619
// MI455X (gfx1250) — matrix operand order repaired and checked by reading, not compiled, not run on hardware
//
#include <hip/hip_runtime.h>
#include <hip/hip_bf16.h>

typedef __attribute__((ext_vector_type(16))) _Float16 v16h;
typedef __attribute__((ext_vector_type(8)))  float    v8f;

#define B_   128
#define F_   8192
#define H_   8192
#define S_   4
#define K_   32
#define L_   3
#define C_   100
#define WORDS_PER_ROW 256   // H_/32 == F_/32
#define LDS_STRIDE    257   // pad: lane b, word w -> bank (b+w)%64, conflict-free
#define ACT_WORDS     (B_ * WORDS_PER_ROW)   // 32768 words per packed activation buffer

// ---------------------------------------------------------------------------
// 1) Bit-pack x (values are 0.0 / 1.0) into 32-bit words using wave32 ballot.
// ---------------------------------------------------------------------------
__global__ void __launch_bounds__(256) eisani_pack(const float* __restrict__ x,
                                                   unsigned* __restrict__ px,
                                                   int nWords) {
  int gwave = blockIdx.x * (blockDim.x >> 5) + (threadIdx.x >> 5);
  int lane  = threadIdx.x & 31;
  if (gwave >= nWords) return;
  float v = x[gwave * 32 + lane];
  unsigned m = __builtin_amdgcn_ballot_w32(v != 0.0f);
  if (lane == 0) px[gwave] = m;
}

// ---------------------------------------------------------------------------
// 2) Segment layer: bit-packed activations for 32 batch rows live in LDS.
//    Wave layout: lane = batch, each wave walks 32 consecutive h and packs
//    the fired-bits into one register word -> single store, no atomics.
//    idx/signs addresses are wave-uniform (readfirstlane) -> broadcast loads.
// ---------------------------------------------------------------------------
__global__ void __launch_bounds__(256) eisani_layer(const unsigned* __restrict__ actIn,
                                                    const int*      __restrict__ idx,
                                                    const float*    __restrict__ sgn,
                                                    const int*      __restrict__ thrPtr,
                                                    unsigned*       __restrict__ actOut) {
  __shared__ unsigned lds[32 * LDS_STRIDE];
  const int bTileBase = blockIdx.y * 32;

  // cooperative fill: 32 rows x 256 words, padded stride in LDS
  for (int i = threadIdx.x; i < 32 * WORDS_PER_ROW; i += 256) {
    int bl = i >> 8, w = i & 255;
    lds[bl * LDS_STRIDE + w] = actIn[(bTileBase + bl) * WORDS_PER_ROW + w];
  }
  __syncthreads();

  const int thr  = *thrPtr;
  const int lane = threadIdx.x & 31;
  const int wid  = threadIdx.x >> 5;
  const int hBase = blockIdx.x * 256 + wid * 32;
  const unsigned* __restrict__ myRow = &lds[lane * LDS_STRIDE];

  unsigned word = 0;
  for (int hi = 0; hi < 32; ++hi) {
    int h = hBase + hi;
    bool fired = false;
#pragma unroll
    for (int s = 0; s < S_; ++s) {
      // wave-uniform base address for the idx/sign stream
      int base = __builtin_amdgcn_readfirstlane((s * H_ + h) * K_);
      __builtin_prefetch(&idx[base + K_], 0, 1);
      __builtin_prefetch(&sgn[base + K_], 0, 1);
      int z = 0;
#pragma unroll
      for (int k = 0; k < K_; ++k) {
        int j = base + k;
        int id = idx[j];
        int neg = (int)(__float_as_uint(sgn[j]) >> 31);    // 1 if sign == -1
        unsigned bit = (myRow[id >> 5] >> (id & 31)) & 1u; // LDS gather
        z += (int)bit * (1 - 2 * neg);
      }
      fired = fired || (z >= thr);
    }
    word |= (unsigned)fired << hi;
  }
  actOut[(bTileBase + lane) * WORDS_PER_ROW + blockIdx.x * 8 + wid] = word;
}

// ---------------------------------------------------------------------------
// 3) Output GEMM: out[b,c] = sum_l sum_h act_l[b,h] * wout[l,h,c]
//    256 threads = 8 waves per 16x16 output tile, split-K across waves
//    (96 of 768 K-steps each) + LDS reduction. Per step: A built from
//    packed activation bits (exact in f16), wout split hi/lo f16 -> two
//    chained v_wmma_f32_16x16x32_f16 with f32 accumulation (fp32-accurate).
//    Edge columns: clamp address + post-select zero -> branch-free loads
//    (v_cndmask) instead of per-load exec-mask branches.
// ---------------------------------------------------------------------------
union V16u { v16h v; _Float16 e[16]; };

#define TOTAL_STEPS (L_ * (H_ / 32))      // 768
#define STEPS_PER_WAVE (TOTAL_STEPS / 8)  // 96

__global__ void __launch_bounds__(256) eisani_gemm(const unsigned* __restrict__ actsBase,
                                                   const float*    __restrict__ wout,
                                                   float*          __restrict__ out) {
  __shared__ float red[8 * 256];

  const int lane  = threadIdx.x & 31;
  const int wid   = threadIdx.x >> 5;   // 0..7: K-chunk owner
  const int bTile = blockIdx.x;         // 0..7  (rows of 16)
  const int cBase = blockIdx.y * 16;    // 0..96 (cols of 16)
  const int rn    = lane & 15;          // row (A) / col (B) index within tile
  const bool hiH  = lane >= 16;

  v8f acc = {};

  const int  kA0 = hiH ? 16 : 0;
  const int  kA1 = hiH ? 24 : 8;
  const int  kB0 = hiH ? 16 : 0;
  const int  c   = cBase + rn;
  const bool inB = (c < C_);
  const int  cc  = inB ? c : (C_ - 1);   // clamped, always in-bounds

  for (int i = 0; i < STEPS_PER_WAVE; ++i) {
    const int t  = wid * STEPS_PER_WAVE + i;   // global K-step 0..767
    const int l  = t >> 8;                     // layer
    const int hb = (t & 255) << 5;             // h base within layer

    // ---- A fragment: 16x32 f16, contiguous k order, same as B:
    // lanes 0-15 (M=rn): VGPR0-3 K=0..7,  VGPR4-7 K=8..15
    // lanes16-31 (M=rn): VGPR0-3 K=16..23, VGPR4-7 K=24..31
    unsigned w = actsBase[l * ACT_WORDS + (bTile * 16 + rn) * WORDS_PER_ROW + (hb >> 5)];
    V16u A;
#pragma unroll
    for (int j = 0; j < 8; ++j) {
      A.e[j]     = ((w >> (kA0 + j)) & 1u) ? (_Float16)1.0f : (_Float16)0.0f;
      A.e[8 + j] = ((w >> (kA1 + j)) & 1u) ? (_Float16)1.0f : (_Float16)0.0f;
    }
    // ---- B fragment: 32x16 f16, lanes 0-15: N=rn, K=0..15; lanes16-31: K=16..31
    // Branch-free: unconditional load at clamped column, then cndmask to zero.
    V16u Bh, Bl;
#pragma unroll
    for (int j = 0; j < 16; ++j) {
      int k = kB0 + j;
      float v = wout[((l * H_) + (hb + k)) * C_ + cc];
      v = inB ? v : 0.0f;
      _Float16 hi = (_Float16)v;
      Bh.e[j] = hi;
      Bl.e[j] = (_Float16)(v - (float)hi);
    }
    acc = __builtin_amdgcn_wmma_f32_16x16x32_f16(false, A.v, false, Bh.v,
                                                 (short)0, acc, false, false);
    acc = __builtin_amdgcn_wmma_f32_16x16x32_f16(false, A.v, false, Bl.v,
                                                 (short)0, acc, false, false);
  }

  // ---- split-K reduction across the 8 waves
#pragma unroll
  for (int r = 0; r < 8; ++r) red[wid * 256 + lane * 8 + r] = acc[r];
  __syncthreads();

  // thread j owns (lane0 = j>>3, r = j&7) of the C/D fragment layout:
  // VGPR r -> M = r (+8 for lanes 16-31), N = lane0 & 15
  const int j     = threadIdx.x;
  const int lane0 = j >> 3;
  const int r     = j & 7;
  float sum = 0.0f;
#pragma unroll
  for (int wv = 0; wv < 8; ++wv) sum += red[wv * 256 + j];

  const int m    = r + ((lane0 >= 16) ? 8 : 0);
  const int col  = cBase + (lane0 & 15);
  if (col < C_) out[(bTile * 16 + m) * C_ + col] = sum;
}

// ---------------------------------------------------------------------------
// d_in order: 0=x(f32) 1=signs0(f32) 2=signsH(f32) 3=wout(f32)
//             4=idx0(i32) 5=idxH(i32) 6=threshold(i32, 1 elem)
// ---------------------------------------------------------------------------
extern "C" void kernel_launch(void* const* d_in, const int* in_sizes, int n_in,
                              void* d_out, int out_size, void* d_ws, size_t ws_size,
                              hipStream_t stream) {
  const float* x      = (const float*)d_in[0];
  const float* signs0 = (const float*)d_in[1];
  const float* signsH = (const float*)d_in[2];
  const float* wout   = (const float*)d_in[3];
  const int*   idx0   = (const int*)d_in[4];
  const int*   idxH   = (const int*)d_in[5];
  const int*   thr    = (const int*)d_in[6];
  float* out = (float*)d_out;

  unsigned* ws = (unsigned*)d_ws;
  unsigned* px = ws;                          // B_*F_/32 = 32768 words
  unsigned* a0 = ws + ACT_WORDS;              // act layer 0..2, contiguous
  unsigned* a1 = ws + 2 * ACT_WORDS;
  unsigned* a2 = ws + 3 * ACT_WORDS;

  // 1) pack x -> bits
  const int nWords = B_ * F_ / 32;            // 32768
  eisani_pack<<<nWords / 8, 256, 0, stream>>>(x, px, nWords);

  // 2) three segment layers (LDS bit-gather)
  dim3 gl(H_ / 256, B_ / 32);                 // (32, 4)
  eisani_layer<<<gl, 256, 0, stream>>>(px, idx0, signs0, thr, a0);
  eisani_layer<<<gl, 256, 0, stream>>>(a0, idxH,            signsH,            thr, a1);
  eisani_layer<<<gl, 256, 0, stream>>>(a1, idxH + S_*H_*K_, signsH + S_*H_*K_, thr, a2);

  // 3) WMMA output GEMM (128x100 from K = 3*8192), split-K over 8 waves
  dim3 gg(B_ / 16, (C_ + 15) / 16);           // (8, 7)
  eisani_gemm<<<gg, 256, 0, stream>>>(a0, wout, out);
}
